// Attention_39487929319718
// MI455X (gfx1250) — matrix operand order repaired and checked by reading, not compiled, not run on hardware
//
#include <hip/hip_runtime.h>

// ---------------------------------------------------------------------------
// Attention (B=4, C=128, H=W=64, HEADS=4, DH=32, MEM=16) for gfx1250 (CDNA5)
// Flash-attention with v_wmma_f32_16x16x32_f16, wave32, online softmax.
// Transposed formulation: S^T = K·Q  puts the softmax (j) axis on the C-layout
// M dimension -> per-lane reductions, register-only P^T->B-layout conversion.
// ---------------------------------------------------------------------------

typedef __attribute__((ext_vector_type(16))) _Float16 v16h;
typedef __attribute__((ext_vector_type(8)))  float    v8f;
typedef __attribute__((ext_vector_type(4)))  float    f4;

union V16H { v16h v; f4 q[2]; };
union V16HU { v16h v; unsigned u[8]; };

static constexpr int NSEQ = 4112;   // HW + MEM
static constexpr int NPAD = 4160;   // padded to multiple of 64 (65 * 64)
static constexpr int HW_  = 4096;

// ds_swizzle: and_mask=0x1f, xor_mask in offset[14:10]; xor 16 = 0x401f
#define SWZ_F(x, imm) __builtin_bit_cast(float, \
    __builtin_amdgcn_ds_swizzle(__builtin_bit_cast(int, (x)), (imm)))

// ---------------------------------------------------------------------------
// Kernel 1: QKV projection.  qkv[o,n] = sum_c w_qkv[o,c] * x_ext[c,n]
//   Q  -> f16 [bh][n][32]      (pre-scaled by DH^-0.5 * log2(e))
//   K  -> f16 [bh][npad][32]   (pad rows zero)
//   Vt -> f16 [bh][32][npad]   (pad cols zero)
// 384 threads: one output row o per thread, 16 columns n per block.
// ---------------------------------------------------------------------------
__global__ __launch_bounds__(384) void
qkv_proj_kernel(const float* __restrict__ x, const float* __restrict__ memv,
                const float* __restrict__ wqkv,
                _Float16* __restrict__ Q, _Float16* __restrict__ K,
                _Float16* __restrict__ V)
{
  __shared__ __align__(16) float xt[128 * 16];   // [c][nl]
  const int b   = blockIdx.y;
  const int n0  = blockIdx.x * 16;               // 0..4144 step 16
  const int t   = threadIdx.x;
  const bool pad = (n0 >= NSEQ);

  if (!pad) {
    if (n0 < HW_) {
      for (int e = t; e < 2048; e += 384) {
        const int c = e >> 4, nl = e & 15;
        xt[e] = x[((size_t)b * 128 + c) * HW_ + n0 + nl];
      }
    } else {                                     // memory tile (n0 == 4096)
      for (int e = t; e < 2048; e += 384) xt[e] = memv[e];
    }
  }
  __syncthreads();

  float acc[16];
#pragma unroll
  for (int i = 0; i < 16; ++i) acc[i] = 0.0f;

  if (!pad) {
    const float* wr = wqkv + (size_t)t * 128;
    for (int c = 0; c < 128; c += 4) {
      const f4  wv = *(const f4*)(wr + c);
      const f4* xr = (const f4*)(xt + c * 16);
#pragma unroll
      for (int k = 0; k < 4; ++k) {
        const float wk = wv[k];
#pragma unroll
        for (int g = 0; g < 4; ++g) {
          const f4 xv = xr[k * 4 + g];           // LDS broadcast across threads
          acc[g * 4 + 0] += wk * xv[0];
          acc[g * 4 + 1] += wk * xv[1];
          acc[g * 4 + 2] += wk * xv[2];
          acc[g * 4 + 3] += wk * xv[3];
        }
      }
    }
  }

  const int o = t;
  if (o < 128) {                                 // Q rows
    if (!pad) {
      const int h = o >> 5, d = o & 31;
      _Float16* qp = Q + (((size_t)(b * 4 + h) * NSEQ) + n0) * 32 + d;
      // fold softmax scale and log2(e) so attention can use exp2 directly
      const float QS = (float)(0.17677669529663687 * 1.4426950408889634);
#pragma unroll
      for (int nl = 0; nl < 16; ++nl) qp[nl * 32] = (_Float16)(acc[nl] * QS);
    }
  } else if (o < 256) {                          // K rows (zeros in pad tiles)
    const int o2 = o - 128, h = o2 >> 5, d = o2 & 31;
    _Float16* kp = K + (((size_t)(b * 4 + h) * NPAD) + n0) * 32 + d;
#pragma unroll
    for (int nl = 0; nl < 16; ++nl) kp[nl * 32] = (_Float16)acc[nl];
  } else {                                       // V rows, stored transposed
    const int o2 = o - 256, h = o2 >> 5, d = o2 & 31;
    _Float16* vq = V + ((size_t)(b * 4 + h) * 32 + d) * NPAD + n0;
#pragma unroll
    for (int nl = 0; nl < 16; ++nl) vq[nl] = (_Float16)acc[nl];
  }
}

// ---------------------------------------------------------------------------
// Kernel 2: flash attention, transposed tiles.  One workgroup = (b,h) x 128
// query columns; 8 waves, each owns 16 i-columns; 65 key chunks of 64.
//
// A-layout (16x32 f16): lane L -> row M=L&15; halves 0..7 = K hiA..hiA+7,
//   halves 8..15 = K 16+hiA..23+hiA, hiA = (L>=16)?8:0.
// B-layout (32x16 f16): lane L -> col N=L&15; Q: K order as A; PV: K 0..15 (L<16) / 16..31 (P^T and Vt).
// C-layout (16x16 f32): vgpr r -> M = r + (L>=16?8:0), N = L&15.
//
// S^T tile t: M = j = jb+16t+r+hoff, N = i  -> softmax stats are per-lane
// scalars (one xor-16 swizzle to merge the partner half-lane's 8 rows).
// P^T -> PV B-layout: pack f16 pairs, one xor-16 swizzle per dword to swap
// half-lane row groups.  O^T accumulated with A = Vt tiles.
// ---------------------------------------------------------------------------
__global__ __launch_bounds__(256) void
attn_flash_kernel(const _Float16* __restrict__ Q, const _Float16* __restrict__ K,
                  const _Float16* __restrict__ V, float* __restrict__ AO)
{
  const int  lane  = threadIdx.x & 31;
  const int  l15   = lane & 15;
  const bool lo16  = (lane & 16) == 0;
  const int  hiA   = lo16 ? 0 : 8;
  const int  hiB   = lo16 ? 0 : 16;
  const int  hoff  = lo16 ? 0 : 8;
  const int  wv_   = threadIdx.x >> 5;
  const int  bh    = blockIdx.y;                 // b*4 + h
  const int  iBase = blockIdx.x * 128 + wv_ * 16;

  // Q B-tile (32d x 16i): fixed for the whole loop
  V16H qb;
  {
    const _Float16* qp = Q + ((size_t)bh * NSEQ + (size_t)(iBase + l15)) * 32 + hiA;
    qb.q[0] = *(const f4*)(qp);
    qb.q[1] = *(const f4*)(qp + 16);
  }

  const _Float16* kp = K + (size_t)bh * NPAD * 32 + (size_t)l15 * 32 + hiA;
  const _Float16* vp = V + ((size_t)bh * 32 + (size_t)l15) * NPAD + hiB;

  float m = -3.0e38f, l = 0.0f;
  v8f O0 = {}, O1 = {};                           // O^T: d = 0..15 / 16..31

  for (int jt = 0; jt < NPAD / 64; ++jt) {
    const int jb = jt * 64;

    // ---- S^T tiles: A = K rows j, B = Q -------------------------------
    v8f S[4];
    const v8f zc = {};
#pragma unroll
    for (int t = 0; t < 4; ++t) {
      V16H ka;
      const _Float16* kpt = kp + (size_t)(jb + 16 * t) * 32;
      ka.q[0] = *(const f4*)(kpt);
      ka.q[1] = *(const f4*)(kpt + 16);
      S[t] = __builtin_amdgcn_wmma_f32_16x16x32_f16(false, ka.v, false, qb.v,
                                                    (short)0, zc, false, false);
    }

    // last chunk: j = 4096..4159; tiles 1..3 (4112..4159) are all padding
    if (jt == NPAD / 64 - 1) {
#pragma unroll
      for (int t = 1; t < 4; ++t)
#pragma unroll
        for (int r = 0; r < 8; ++r) S[t][r] = -3.0e38f;
    }

    // ---- online softmax: per-lane column stats ------------------------
    float cmax;
    {
      float a0 = fmaxf(S[0][0], S[0][1]), a1 = fmaxf(S[0][2], S[0][3]);
      float a2 = fmaxf(S[0][4], S[0][5]), a3 = fmaxf(S[0][6], S[0][7]);
#pragma unroll
      for (int t = 1; t < 4; ++t) {
        a0 = fmaxf(a0, fmaxf(S[t][0], S[t][1]));
        a1 = fmaxf(a1, fmaxf(S[t][2], S[t][3]));
        a2 = fmaxf(a2, fmaxf(S[t][4], S[t][5]));
        a3 = fmaxf(a3, fmaxf(S[t][6], S[t][7]));
      }
      cmax = fmaxf(fmaxf(a0, a1), fmaxf(a2, a3));
    }
    cmax = fmaxf(cmax, SWZ_F(cmax, 0x401f));     // merge partner half-lane

    const float mn    = fmaxf(m, cmax);
    const float alpha = __builtin_amdgcn_exp2f(m - mn);  // S in log2 domain
    m = mn;

    float csum = 0.0f;
#pragma unroll
    for (int t = 0; t < 4; ++t)
#pragma unroll
      for (int r = 0; r < 8; ++r) {
        S[t][r] = __builtin_amdgcn_exp2f(S[t][r] - mn);
        csum += S[t][r];
      }
    csum += SWZ_F(csum, 0x401f);
    l = l * alpha + csum;

#pragma unroll
    for (int r = 0; r < 8; ++r) { O0[r] *= alpha; O1[r] *= alpha; }

    // ---- P^T -> B-layout in registers, accumulate O^T -----------------
#pragma unroll
    for (int c = 0; c < 2; ++c) {                // two 32-j chunks
      V16HU pb;
#pragma unroll
      for (int r4 = 0; r4 < 4; ++r4) {
        const unsigned plo = __builtin_bit_cast(unsigned,
            __builtin_amdgcn_cvt_pkrtz(S[2 * c][2 * r4], S[2 * c][2 * r4 + 1]));
        const unsigned phi = __builtin_bit_cast(unsigned,
            __builtin_amdgcn_cvt_pkrtz(S[2 * c + 1][2 * r4], S[2 * c + 1][2 * r4 + 1]));
        const unsigned send = lo16 ? phi : plo;  // give partner its missing rows
        const unsigned recv = (unsigned)__builtin_amdgcn_ds_swizzle((int)send, 0x401f);
        pb.u[r4]     = lo16 ? plo : recv;        // chunk rows 0..7  of my tile
        pb.u[4 + r4] = lo16 ? recv : phi;        // chunk rows 8..15 of my tile
      }

      V16H va0, va1;                             // A = Vt rows d, K-dim = j
      const _Float16* vpt = vp + jb + 32 * c;
      va0.q[0] = *(const f4*)(vpt);
      va0.q[1] = *(const f4*)(vpt + 8);
      va1.q[0] = *(const f4*)(vpt + 16 * NPAD);
      va1.q[1] = *(const f4*)(vpt + 16 * NPAD + 8);

      O0 = __builtin_amdgcn_wmma_f32_16x16x32_f16(false, va0.v, false, pb.v,
                                                  (short)0, O0, false, false);
      O1 = __builtin_amdgcn_wmma_f32_16x16x32_f16(false, va1.v, false, pb.v,
                                                  (short)0, O1, false, false);
    }
  }

  // ---- normalize, store AO[b*128 + h*32 + d][i] ------------------------
  const float inv = 1.0f / l;
  float* ao = AO + (size_t)(bh * 32) * HW_ + (size_t)(iBase + l15);
#pragma unroll
  for (int r = 0; r < 8; ++r) {
    ao[(size_t)(r + hoff) * HW_]      = O0[r] * inv;
    ao[(size_t)(16 + r + hoff) * HW_] = O1[r] * inv;
  }
}

// ---------------------------------------------------------------------------
// Kernel 3: output projection  out[b,o,n] = sum_c w_out[o,c]*AO[b,c,n] + b_out[o]
// ---------------------------------------------------------------------------
__global__ __launch_bounds__(128) void
out_proj_kernel(const float* __restrict__ AO, const float* __restrict__ wout,
                const float* __restrict__ bout, float* __restrict__ out)
{
  __shared__ __align__(16) float at[128 * 16];
  const int b  = blockIdx.y;
  const int n0 = blockIdx.x * 16;
  const int t  = threadIdx.x;

  for (int e = t; e < 2048; e += 128) {
    const int c = e >> 4, nl = e & 15;
    at[e] = AO[((size_t)b * 128 + c) * HW_ + n0 + nl];
  }
  __syncthreads();

  float acc[16];
  const float bb = bout[t];
#pragma unroll
  for (int i = 0; i < 16; ++i) acc[i] = bb;

  const float* wr = wout + (size_t)t * 128;
  for (int c = 0; c < 128; c += 4) {
    const f4  wv = *(const f4*)(wr + c);
    const f4* xr = (const f4*)(at + c * 16);
#pragma unroll
    for (int k = 0; k < 4; ++k) {
      const float wk = wv[k];
#pragma unroll
      for (int g = 0; g < 4; ++g) {
        const f4 xv = xr[k * 4 + g];
        acc[g * 4 + 0] += wk * xv[0];
        acc[g * 4 + 1] += wk * xv[1];
        acc[g * 4 + 2] += wk * xv[2];
        acc[g * 4 + 3] += wk * xv[3];
      }
    }
  }

  float* op = out + ((size_t)b * 128 + t) * HW_ + n0;
#pragma unroll
  for (int nl = 0; nl < 16; ++nl) op[nl] = acc[nl];
}

// ---------------------------------------------------------------------------
// Workspace layout (bytes):
//   Q : 16 * 4112 * 32 * 2 = 4,210,688
//   K : 16 * 4160 * 32 * 2 = 4,259,840
//   Vt: 16 * 4160 * 32 * 2 = 4,259,840
//   AO: 4 * 128 * 4096 * 4 = 8,388,608     total ~21.1 MB
// ---------------------------------------------------------------------------
extern "C" void kernel_launch(void* const* d_in, const int* in_sizes, int n_in,
                              void* d_out, int out_size, void* d_ws, size_t ws_size,
                              hipStream_t stream)
{
  (void)in_sizes; (void)n_in; (void)out_size; (void)ws_size;

  const float* x    = (const float*)d_in[0];
  const float* memv = (const float*)d_in[1];
  const float* wqkv = (const float*)d_in[2];
  const float* wout = (const float*)d_in[3];
  const float* bout = (const float*)d_in[4];
  float*       out  = (float*)d_out;

  char* ws = (char*)d_ws;
  _Float16* Q  = (_Float16*)(ws);
  _Float16* K  = (_Float16*)(ws + 4210688);
  _Float16* Vt = (_Float16*)(ws + 8470528);
  float*    AO = (float*)   (ws + 12730368);

  qkv_proj_kernel  <<<dim3(NPAD / 16, 4), 384, 0, stream>>>(x, memv, wqkv, Q, K, Vt);
  attn_flash_kernel<<<dim3(HW_ / 128, 16), 256, 0, stream>>>(Q, K, Vt, AO);
  out_proj_kernel  <<<dim3(HW_ / 16, 4), 128, 0, stream>>>(AO, wout, bout, out);
}
